// CrossLayerTranscoder_52604759441480
// MI455X (gfx1250) — compile-verified, matrix operand order checked by reading, not run on hardware
//
#include <hip/hip_runtime.h>

// Problem constants (from reference)
#define D_IN   2048
#define D_HID  16384
#define D_OUT  2048
#define KTOP   64
#define NTOK   16384      // B*S = 8*2048
#define CHUNK  1024       // tokens per encoder/top-k pass

// Encoder GEMM tiling
#define MT 32             // tokens per block (and per wave)
#define NT 512            // hidden units per block (64 per wave x 8 waves)
#define KT 64             // k elements per LDS stage
#define TSTAGES (D_IN / KT)

typedef __attribute__((ext_vector_type(16))) __bf16  v16bf;
typedef __attribute__((ext_vector_type(8)))  float   v8f;
typedef unsigned int v4u __attribute__((ext_vector_type(4)));
typedef int          v8i __attribute__((ext_vector_type(8)));
typedef int          v4i __attribute__((ext_vector_type(4)));

__device__ __forceinline__ unsigned short f32_to_bf16_bits(float f) {
  unsigned int u = __float_as_uint(f);
  unsigned int r = u + 0x7FFFu + ((u >> 16) & 1u);   // round-to-nearest-even
  return (unsigned short)(r >> 16);
}
__device__ __forceinline__ float bf16_bits_to_f32(unsigned short h) {
  return __uint_as_float(((unsigned int)h) << 16);
}

// ---- Tensor Data Mover: 2D bf16 tile (tile_d1 rows x tile_d0 elems) -> LDS ----
// D# packing per CDNA5 ISA (2-D tensor: groups 2/3 zero).
// 6-arg builtin form (amdgpu-toolchain / clang-23).
__device__ __forceinline__ void tdm_load_2d_bf16(unsigned lds_off, const void* gaddr,
                                                 unsigned tensor_d0, unsigned tensor_d1,
                                                 unsigned tile_d0, unsigned tile_d1,
                                                 unsigned stride_d0_elems) {
  unsigned long long ga = (unsigned long long)(uintptr_t)gaddr;
  v4u g0;
  g0[0] = 1u;                                            // count=1, user mode
  g0[1] = lds_off;                                       // lds_addr (bytes)
  g0[2] = (unsigned)(ga & 0xFFFFFFFFu);                  // global_addr[31:0]
  g0[3] = (unsigned)((ga >> 32) & 0x1FFFFFFu)            // global_addr[56:32]
        | (2u << 30);                                    // type = 2 ("image")
  v8i g1;
  g1[0] = (int)(1u << 16);                               // data_size=1 (2 bytes), mask=0
  g1[1] = (int)((tensor_d0 & 0xFFFFu) << 16);            // tensor_dim0[15:0] @ bit48
  g1[2] = (int)((tensor_d0 >> 16) | ((tensor_d1 & 0xFFFFu) << 16)); // dim0 hi | dim1 lo
  g1[3] = (int)((tensor_d1 >> 16) | (tile_d0 << 16));    // dim1 hi | tile_dim0
  g1[4] = (int)(tile_d1 & 0xFFFFu);                      // tile_dim1 (tile_dim2 = 0)
  g1[5] = (int)stride_d0_elems;                          // tensor_dim0_stride[31:0]
  g1[6] = 0;                                             // stride hi | dim1_stride lo
  g1[7] = 0;
  v4i z4 = {0, 0, 0, 0};
  v8i z8 = {0, 0, 0, 0, 0, 0, 0, 0};
  __builtin_amdgcn_tensor_load_to_lds(g0, g1, z4, z4, z8, 0);
}

// ---------------- fp32 -> bf16 cast (4 elems/thread) ----------------
__global__ void __launch_bounds__(256)
cast_bf16_kernel(const float* __restrict__ src,
                 unsigned short* __restrict__ dst, int n) {
  int base = (blockIdx.x * blockDim.x + threadIdx.x) * 4;
  if (base + 3 < n) {
    float4 v = *(const float4*)(src + base);
    ushort4 o;
    o.x = f32_to_bf16_bits(v.x);
    o.y = f32_to_bf16_bits(v.y);
    o.z = f32_to_bf16_bits(v.z);
    o.w = f32_to_bf16_bits(v.w);
    *(ushort4*)(dst + base) = o;
  }
}

// ---------------- W_dec (D_OUT x D_HID) -> W_decT bf16 (D_HID x D_OUT) ----------------
__global__ void __launch_bounds__(256)
transpose_wdec_kernel(const float* __restrict__ Wdec,
                      unsigned short* __restrict__ WdecT) {
  __shared__ float tile[32][33];
  const int tx = threadIdx.x;   // 0..31
  const int ty = threadIdx.y;   // 0..7
  const int h0 = blockIdx.x * 32;
  const int o0 = blockIdx.y * 32;
#pragma unroll
  for (int i = 0; i < 4; ++i) {
    int o = o0 + ty + i * 8;
    tile[ty + i * 8][tx] = Wdec[(size_t)o * D_HID + h0 + tx];
  }
  __syncthreads();
#pragma unroll
  for (int i = 0; i < 4; ++i) {
    int h = h0 + ty + i * 8;
    WdecT[(size_t)h * D_OUT + o0 + tx] = f32_to_bf16_bits(tile[tx][ty + i * 8]);
  }
}

// ---------------- Encoder GEMM: feats = relu(x @ W_enc^T + b_enc - thr) ----------------
// 8 waves; block tile 32 tokens x 512 hidden; each wave computes 32x64
// (8 v8f accumulators). TDM double-buffers A (32x64) and B (512x64) bf16
// tiles in LDS (136 KB), wave 0 drives the DMA (TENSORcnt); each B fragment
// read from LDS feeds two WMMAs (1.33 WMMA per ds fragment).
__global__ void __launch_bounds__(256)
enc_gemm_kernel(const unsigned short* __restrict__ Xb,   // CHUNK x D_IN   (bf16 bits)
                const unsigned short* __restrict__ Wb,   // D_HID x D_IN   (bf16 bits)
                const float* __restrict__ b_enc,
                const float* __restrict__ thr,
                float* __restrict__ feats)               // CHUNK x D_HID  (fp32)
{
  __shared__ __align__(16) unsigned short sA[2][MT * KT];   // 2 x 4 KB
  __shared__ __align__(16) unsigned short sB[2][NT * KT];   // 2 x 64 KB

  const int tid  = threadIdx.x;
  const int lane = tid & 31;
  const int wave = tid >> 5;           // 0..7 -> N sub-tile

  const int mBase = blockIdx.y * MT;   // token tile within chunk
  const int hBase = blockIdx.x * NT;   // hidden tile

  const int lrow = lane & 15;
  const int kh   = (lane >> 4) * 16;

  v8f acc[2][4];
#pragma unroll
  for (int ms = 0; ms < 2; ++ms)
#pragma unroll
    for (int t = 0; t < 4; ++t) acc[ms][t] = (v8f){};

  if (wave == 0) {
    // prologue: stage 0 into buffer 0
    tdm_load_2d_bf16((unsigned)(uintptr_t)&sA[0][0], Xb + (size_t)mBase * D_IN,
                     D_IN, CHUNK, KT, MT, D_IN);
    tdm_load_2d_bf16((unsigned)(uintptr_t)&sB[0][0], Wb + (size_t)hBase * D_IN,
                     D_IN, D_HID, KT, NT, D_IN);
  }

  for (int t = 0; t < TSTAGES; ++t) {
    const int buf = t & 1;
    if (wave == 0) {
      if (t + 1 < TSTAGES) {
        const unsigned k = (unsigned)(t + 1) * KT;
        tdm_load_2d_bf16((unsigned)(uintptr_t)&sA[buf ^ 1][0],
                         Xb + (size_t)mBase * D_IN + k, D_IN, CHUNK, KT, MT, D_IN);
        tdm_load_2d_bf16((unsigned)(uintptr_t)&sB[buf ^ 1][0],
                         Wb + (size_t)hBase * D_IN + k, D_IN, D_HID, KT, NT, D_IN);
        // next stage (2 tensor ops) may remain in flight; current stage done
        __builtin_amdgcn_s_wait_tensorcnt((short)2);
      } else {
        __builtin_amdgcn_s_wait_tensorcnt((short)0);
      }
    }
    __syncthreads();    // publish stage t to all waves

#pragma unroll
    for (int ks = 0; ks < 2; ++ks) {
      const unsigned short* pa = &sA[buf][lrow * KT + ks * 32 + kh];
      const unsigned short* pb = &sB[buf][(wave * 64 + lrow) * KT + ks * 32 + kh];
      v16bf a0 = *(const v16bf*)(pa);             // tokens 0..15
      v16bf a1 = *(const v16bf*)(pa + 16 * KT);   // tokens 16..31
#pragma unroll
      for (int nt = 0; nt < 4; ++nt) {
        v16bf b = *(const v16bf*)(pb + nt * 16 * KT);
        acc[0][nt] = __builtin_amdgcn_wmma_f32_16x16x32_bf16(false, a0, false, b, (short)0, acc[0][nt], false, false);
        acc[1][nt] = __builtin_amdgcn_wmma_f32_16x16x32_bf16(false, a1, false, b, (short)0, acc[1][nt], false, false);
      }
    }
    __syncthreads();    // all waves done reading before buf^1 gets overwritten
  }

  // Epilogue: C layout — lanes = N (0..15 twice), VGPR r: M = r + (lane>>4)*8
  const int n    = lane & 15;
  const int mOff = (lane >> 4) * 8;
#pragma unroll
  for (int nt = 0; nt < 4; ++nt) {
    const int h = hBase + wave * 64 + nt * 16 + n;
    const float c = b_enc[h] - thr[h];
#pragma unroll
    for (int ms = 0; ms < 2; ++ms) {
#pragma unroll
      for (int r = 0; r < 8; ++r) {
        const size_t row = (size_t)(mBase + ms * 16 + mOff + r) * D_HID;
        feats[row + h] = fmaxf(acc[ms][nt][r] + c, 0.0f);
      }
    }
  }
}

// ---------------- Top-K per token (block = 256 threads, token row in registers) ----------------
__global__ void __launch_bounds__(256)
topk_kernel(const float* __restrict__ feats,   // CHUNK x D_HID
            float* __restrict__ vals,          // (chunk-offset) CHUNK x KTOP
            int*   __restrict__ idx)
{
  const int tok = blockIdx.x;
  const int tid = threadIdx.x;
  const float* row = feats + (size_t)tok * D_HID;

  float f[64];
#pragma unroll
  for (int j = 0; j < 64; ++j) f[j] = row[tid + j * 256];

  __shared__ float sv[256];
  __shared__ int   si[256];

  for (int r = 0; r < KTOP; ++r) {
    float bv = -3.4e38f; int bj = 0;
#pragma unroll
    for (int j = 0; j < 64; ++j) {
      if (f[j] > bv) { bv = f[j]; bj = j; }
    }
    sv[tid] = bv;
    si[tid] = tid + bj * 256;
    __syncthreads();
#pragma unroll
    for (int s = 128; s > 0; s >>= 1) {
      if (tid < s) {
        float ov = sv[tid + s]; int oi = si[tid + s];
        if (ov > sv[tid] || (ov == sv[tid] && oi < si[tid])) { sv[tid] = ov; si[tid] = oi; }
      }
      __syncthreads();
    }
    const int   widx = si[0];
    const float wval = sv[0];
    if (tid == 0) { vals[tok * KTOP + r] = wval; idx[tok * KTOP + r] = widx; }
    __syncthreads();                      // everyone done reading sv/si[0]
    if ((widx & 255) == tid) {            // disable winner (unrolled, no dyn index)
      const int jj = widx >> 8;
#pragma unroll
      for (int j = 0; j < 64; ++j) if (j == jj) f[j] = -3.4e38f;
    }
  }
}

// ---------------- Sparse decode: out = sum_k vals*W_decT[idx] + b_dec ----------------
__global__ void __launch_bounds__(256)
decode_kernel(const float* __restrict__ vals,          // NTOK x KTOP
              const int*   __restrict__ idx,
              const unsigned short* __restrict__ WdT,  // D_HID x D_OUT (bf16 bits)
              const float* __restrict__ b_dec,
              float* __restrict__ out)                 // NTOK x D_OUT
{
  const int tok = blockIdx.x;
  const int tid = threadIdx.x;
  __shared__ float lv[KTOP];
  __shared__ int   li[KTOP];
  if (tid < KTOP) { lv[tid] = vals[tok * KTOP + tid]; li[tid] = idx[tok * KTOP + tid]; }
  __syncthreads();

  float acc[8];
#pragma unroll
  for (int j = 0; j < 8; ++j) acc[j] = b_dec[tid + j * 256];

  for (int t = 0; t < KTOP; ++t) {
    const float v = lv[t];
    const unsigned short* wrow = WdT + (size_t)li[t] * D_OUT;
#pragma unroll
    for (int j = 0; j < 8; ++j)
      acc[j] = fmaf(v, bf16_bits_to_f32(wrow[tid + j * 256]), acc[j]);
  }
#pragma unroll
  for (int j = 0; j < 8; ++j) out[(size_t)tok * D_OUT + tid + j * 256] = acc[j];
}

extern "C" void kernel_launch(void* const* d_in, const int* in_sizes, int n_in,
                              void* d_out, int out_size, void* d_ws, size_t ws_size,
                              hipStream_t stream) {
  const float* x     = (const float*)d_in[0];
  const float* W_enc = (const float*)d_in[1];
  const float* b_enc = (const float*)d_in[2];
  const float* thr   = (const float*)d_in[3];
  const float* W_dec = (const float*)d_in[4];
  const float* b_dec = (const float*)d_in[5];
  float* out = (float*)d_out;

  // Workspace layout (64 MB sections; total ~264 MB)
  char* ws = (char*)d_ws;
  const size_t SZ = (size_t)NTOK * D_IN * 2;                 // 64 MB
  unsigned short* xb    = (unsigned short*)(ws);             // x bf16
  unsigned short* web   = (unsigned short*)(ws + SZ);        // W_enc bf16
  unsigned short* wdt   = (unsigned short*)(ws + 2 * SZ);    // W_dec^T bf16
  float*          feats = (float*)(ws + 3 * SZ);             // CHUNK x D_HID fp32
  float*          vals  = (float*)(ws + 4 * SZ);             // NTOK x KTOP
  int*            idxb  = (int*)(ws + 4 * SZ + (size_t)NTOK * KTOP * 4);

  // Casts
  {
    const int n = NTOK * D_IN;          // == D_HID * D_IN == 33,554,432
    cast_bf16_kernel<<<n / (256 * 4), 256, 0, stream>>>(x, xb, n);
    cast_bf16_kernel<<<n / (256 * 4), 256, 0, stream>>>(W_enc, web, n);
  }
  transpose_wdec_kernel<<<dim3(D_HID / 32, D_OUT / 32), dim3(32, 8), 0, stream>>>(W_dec, wdt);

  // Encoder + top-k, CHUNK tokens at a time
  for (int c = 0; c < NTOK / CHUNK; ++c) {
    const unsigned short* xc = xb + (size_t)c * CHUNK * D_IN;
    enc_gemm_kernel<<<dim3(D_HID / NT, CHUNK / MT), 256, 0, stream>>>(
        xc, web, b_enc, thr, feats);
    topk_kernel<<<CHUNK, 256, 0, stream>>>(
        feats, vals + (size_t)c * CHUNK * KTOP, idxb + (size_t)c * CHUNK * KTOP);
  }

  // Sparse decode (W_dec^T bf16 = 64 MB, L2-resident)
  decode_kernel<<<NTOK, 256, 0, stream>>>(vals, idxb, wdt, b_dec, out);
}
